// Transformer_54571854463216
// MI455X (gfx1250) — matrix operand order repaired and checked by reading, not compiled, not run on hardware
//
#include <hip/hip_runtime.h>

#define DEV __device__ __forceinline__

typedef _Float16 h16;
typedef __attribute__((ext_vector_type(16))) _Float16     v16h;
typedef __attribute__((ext_vector_type(8)))  float        v8f;
typedef __attribute__((ext_vector_type(4)))  unsigned int u32x4;

union FragAB {
    v16h h;
    u32x4 q[2];
    unsigned int u[8];
    _Float16 e[16];
};

constexpr int   Bb   = 2, S = 1024, Dm = 1024, H = 16, HD = 64;
constexpr int   Mrow = Bb * S;                 // 2048 GEMM rows
constexpr float NEGV = -4294967295.0f;         // -(2^32 - 1)
constexpr float RATE = 0.1f;
constexpr float EPSf = 1e-8f;
constexpr int   NLAYER = 6;

// --- A-fragment K index for V_WMMA_F32_16X16X32_F16 (ISA 7.12.2, 16-bit A 16x32) ---
// halves per lane form two contiguous 16B runs: [8g..8g+7] and [16+8g..16+8g+7]
DEV constexpr int a_kk(int v, int g) {
    return (v < 4 ? 2 * v : 16 + 2 * (v - 4)) + 8 * g;
}

DEV v8f wmma_f16(const FragAB& a, const FragAB& b, v8f c) {
    return __builtin_amdgcn_wmma_f32_16x16x32_f16(
        false, a.h, false, b.h, (short)0, c, false, false);
}

// ======================= weight packing (f32 -> f16 B-fragment layout) ===============
// Packed layout: halfIdx = frag*512 + lane*16 + idx, frag = kt*(N/16)+nt,
// element (k = kt*32 + (lane>>4)*16 + 2*(idx>>1) + (idx&1), n = nt*16 + (lane&15)).
__global__ void pack_w_qkv(const float* __restrict__ W, h16* __restrict__ out) {
    int tid  = blockIdx.x * 256 + threadIdx.x;      // D*D total
    int frag = tid >> 9, rem = tid & 511;
    int lane = rem >> 4, idx = rem & 15;
    int v = idx >> 1, j = idx & 1;
    int g = lane >> 4, nc = lane & 15;
    int kt = frag >> 6, nt = frag & 63;             // N/16 = 64
    int k = kt * 32 + g * 16 + 2 * v + j;
    int n = nt * 16 + nc;
    int hh = n >> 6, e = n & 63;                    // W is [H, D, HD]
    out[tid] = (h16)W[((size_t)hh * Dm + k) * HD + e];
}

__global__ void pack_w_dd(const float* __restrict__ W, h16* __restrict__ out) {
    int tid  = blockIdx.x * 256 + threadIdx.x;
    int frag = tid >> 9, rem = tid & 511;
    int lane = rem >> 4, idx = rem & 15;
    int v = idx >> 1, j = idx & 1;
    int g = lane >> 4, nc = lane & 15;
    int kt = frag >> 6, nt = frag & 63;
    int k = kt * 32 + g * 16 + 2 * v + j;
    int n = nt * 16 + nc;
    out[tid] = (h16)W[(size_t)k * Dm + n];
}

// ======================= positional encode + pad mask ===============================
__global__ void pos_encode(const float* __restrict__ q, float* __restrict__ x) {
    int row = blockIdx.x;                 // b*S + s
    int s   = row & (S - 1);
    const float* qr = q + (size_t)row * Dm;
    float*       xr = x + (size_t)row * Dm;
    __shared__ float red[256];
    float a = 0.f;
    for (int d = threadIdx.x; d < Dm; d += 256) a += fabsf(qr[d]);
    red[threadIdx.x] = a; __syncthreads();
    for (int o = 128; o > 0; o >>= 1) {
        if (threadIdx.x < o) red[threadIdx.x] += red[threadIdx.x + o];
        __syncthreads();
    }
    float mask = (red[0] > 0.f) ? 1.f : 0.f;
    float pos  = (float)(s + 1);
    const float ln1e5 = 11.512925464970229f;        // log(100000)
    for (int d = threadIdx.x; d < Dm; d += 256) {
        int   i   = d >> 1;
        float div = __expf(-(2.0f * (float)i / (float)Dm) * ln1e5);
        float ang = pos * div;
        float pe  = (d & 1) ? __cosf(ang) : __sinf(ang);
        xr[d] = qr[d] + RATE * pe * mask;
    }
}

__global__ void make_pad(const float* __restrict__ k, float* __restrict__ pad) {
    int row = blockIdx.x;
    const float* kr = k + (size_t)row * Dm;
    __shared__ float red[256];
    float a = 0.f;
    for (int d = threadIdx.x; d < Dm; d += 256) a += fabsf(kr[d]);
    red[threadIdx.x] = a; __syncthreads();
    for (int o = 128; o > 0; o >>= 1) {
        if (threadIdx.x < o) red[threadIdx.x] += red[threadIdx.x + o];
        __syncthreads();
    }
    if (threadIdx.x == 0) pad[row] = (red[0] == 0.f) ? NEGV : 0.f;
}

// ======================= layernorm (torch std, ddof=1) ==============================
__global__ void layernorm(const float* __restrict__ x,
                          const float* __restrict__ alpha, const float* __restrict__ beta,
                          float* __restrict__ y32, h16* __restrict__ y16) {
    int row = blockIdx.x;
    const float* xr = x + (size_t)row * Dm;
    __shared__ float rs[256], rq[256];
    float s = 0.f, q = 0.f;
    for (int d = threadIdx.x; d < Dm; d += 256) { float v = xr[d]; s += v; q += v * v; }
    rs[threadIdx.x] = s; rq[threadIdx.x] = q; __syncthreads();
    for (int o = 128; o > 0; o >>= 1) {
        if (threadIdx.x < o) { rs[threadIdx.x] += rs[threadIdx.x + o];
                               rq[threadIdx.x] += rq[threadIdx.x + o]; }
        __syncthreads();
    }
    float mu  = rs[0] / (float)Dm;
    float var = fmaxf((rq[0] - (float)Dm * mu * mu) / (float)(Dm - 1), 0.f);
    float inv = 1.f / (sqrtf(var) + EPSf);
    float A = alpha[0], Bp = beta[0];
    for (int d = threadIdx.x; d < Dm; d += 256) {
        float v = A * (xr[d] - mu) * inv + Bp;
        if (y32) y32[(size_t)row * Dm + d] = v;
        y16[(size_t)row * Dm + d] = (h16)v;
    }
}

// ======================= WMMA GEMM, fused epilogues ================================
enum { EP_QS = 0, EP_QK = 1, EP_VT = 2, EP_RES = 3, EP_RELU = 4 };

template <int MODE>
__global__ __launch_bounds__(256)
void gemm_wmma(const h16* __restrict__ A, const h16* __restrict__ Bpk,
               const float* __restrict__ bias, const float* __restrict__ res,
               void* __restrict__ dst) {
    const int lane = threadIdx.x & 31, wave = threadIdx.x >> 5;
    const int g = lane >> 4, hn = lane & 15;
    const int rowBase = blockIdx.y * 256 + wave * 32;    // 8 waves x 32 rows
    const int colBase = blockIdx.x * 64;                 // 4 n-tiles per wave

    v8f zero = {};
    v8f acc[2][4];
    #pragma unroll
    for (int i = 0; i < 2; ++i)
        #pragma unroll
        for (int t = 0; t < 4; ++t) acc[i][t] = zero;

    for (int kt = 0; kt < Dm / 32; ++kt) {
        FragAB af[2];
        #pragma unroll
        for (int sub = 0; sub < 2; ++sub) {
            const h16* arow = A + (size_t)(rowBase + sub * 16 + hn) * Dm + kt * 32;
            af[sub].q[0] = *(const u32x4*)(arow + 16 * g);        // halves [16g..16g+7]
            af[sub].q[1] = *(const u32x4*)(arow + 16 * g + 8);   // halves [16g+8..+7]
        }
        #pragma unroll
        for (int t = 0; t < 4; ++t) {
            int frag = kt * (Dm / 16) + (colBase >> 4) + t;
            FragAB bf;
            bf.h = *(const v16h*)(Bpk + (size_t)frag * 512 + lane * 16);
            #pragma unroll
            for (int sub = 0; sub < 2; ++sub)
                acc[sub][t] = wmma_f16(af[sub], bf, acc[sub][t]);
        }
    }

    #pragma unroll
    for (int sub = 0; sub < 2; ++sub)
        #pragma unroll
        for (int t = 0; t < 4; ++t)
            #pragma unroll
            for (int v = 0; v < 8; ++v) {
                int r = rowBase + sub * 16 + v + 8 * g;     // C layout: row = vgpr + 8*(lane>>4)
                int c = colBase + t * 16 + hn;
                float val = acc[sub][t][v] + bias[c];
                if constexpr (MODE == EP_QS) {              // Q: pre-scale by 1/sqrt(D)
                    int b = r >> 10, s = r & 1023, hh = c >> 6, e = c & 63;
                    ((h16*)dst)[(((size_t)(b * H + hh) * S + s) << 6) + e] =
                        (h16)(val * 0.03125f);
                } else if constexpr (MODE == EP_QK) {
                    int b = r >> 10, s = r & 1023, hh = c >> 6, e = c & 63;
                    ((h16*)dst)[(((size_t)(b * H + hh) * S + s) << 6) + e] = (h16)val;
                } else if constexpr (MODE == EP_VT) {
                    int b = r >> 10, s = r & 1023, hh = c >> 6, e = c & 63;
                    ((h16*)dst)[(((size_t)(b * H + hh) * HD + e) << 10) + s] = (h16)val;
                } else if constexpr (MODE == EP_RES) {
                    ((float*)dst)[(size_t)r * Dm + c] = res[(size_t)r * Dm + c] + val;
                } else {
                    ((float*)dst)[(size_t)r * Dm + c] =
                        res[(size_t)r * Dm + c] + fmaxf(val, 0.f);
                }
            }
}

// ======================= flash attention (online softmax, WMMA) =====================
__global__ __launch_bounds__(128)
void attention(const h16* __restrict__ Q, const h16* __restrict__ K,
               const h16* __restrict__ Vt, const float* __restrict__ pad,
               h16* __restrict__ heads) {
    __shared__ _Float16 probs[4][16][32];              // per-wave P-tile staging
    const int lane = threadIdx.x & 31, wave = threadIdx.x >> 5;
    const int g = lane >> 4, hn = lane & 15;
    const int bh = blockIdx.x >> 4;                    // S/64 = 16 q-tiles per (b,h)
    const int b  = bh >> 4;
    const int hh = bh & 15;
    const int q0 = (blockIdx.x & 15) * 64 + wave * 16;
    const size_t baseQK = (size_t)bh * S * HD;
    const size_t baseVT = (size_t)bh * HD * S;

    FragAB qa[2];                                      // Q A-frags, resident all loop
    {
        const h16* qrow = Q + baseQK + (size_t)(q0 + hn) * HD;
        #pragma unroll
        for (int ch = 0; ch < 2; ++ch) {
            qa[ch].q[0] = *(const u32x4*)(qrow + ch * 32 + 16 * g);
            qa[ch].q[1] = *(const u32x4*)(qrow + ch * 32 + 16 * g + 8);
        }
    }
    FragAB ones;                                       // B-frag of 1.0h for row sums
    #pragma unroll
    for (int i = 0; i < 8; ++i) ones.u[i] = 0x3C003C00u;

    v8f zero = {};
    v8f O[4];
    #pragma unroll
    for (int t = 0; t < 4; ++t) O[t] = zero;
    float mrun = -INFINITY;                            // wave-shared shift (exact softmax)
    float lrun[8];
    #pragma unroll
    for (int v = 0; v < 8; ++v) lrun[v] = 0.f;

    for (int k0 = 0; k0 < S; k0 += 32) {
        v8f s0 = zero, s1 = zero;
        #pragma unroll
        for (int kh = 0; kh < 2; ++kh) {               // two 16-key score tiles
            #pragma unroll
            for (int ch = 0; ch < 2; ++ch) {           // HD=64 contraction: 2 x K32
                FragAB kb;                             // contiguous 32B B-frag
                kb.h = *(const v16h*)(K + baseQK +
                        (size_t)(k0 + kh * 16 + hn) * HD + ch * 32 + 16 * g);
                if (kh == 0) s0 = wmma_f16(qa[ch], kb, s0);
                else         s1 = wmma_f16(qa[ch], kb, s1);
            }
        }
        float pad0 = pad[b * S + k0 + hn];
        float pad1 = pad[b * S + k0 + 16 + hn];
        float mcur = -INFINITY;
        #pragma unroll
        for (int v = 0; v < 8; ++v) {
            s0[v] += pad0;                             // Q pre-scaled by 1/sqrt(D)
            s1[v] += pad1;
            mcur = fmaxf(mcur, fmaxf(s0[v], s1[v]));
        }
        #pragma unroll
        for (int o = 1; o < 32; o <<= 1)               // wave-wide max (5 permutes)
            mcur = fmaxf(mcur, __shfl_xor(mcur, o, 32));
        float mnew = fmaxf(mrun, mcur);
        if (mnew > mrun) {                             // wave-uniform rescale (rare)
            float corr = __expf(mrun - mnew);
            #pragma unroll
            for (int t = 0; t < 4; ++t)
                #pragma unroll
                for (int v = 0; v < 8; ++v) O[t][v] *= corr;
            #pragma unroll
            for (int v = 0; v < 8; ++v) lrun[v] *= corr;
            mrun = mnew;
        }
        #pragma unroll
        for (int v = 0; v < 8; ++v) {                  // C layout: row = v + 8g, col = hn
            probs[wave][v + 8 * g][hn]      = (h16)__expf(s0[v] - mrun);
            probs[wave][v + 8 * g][16 + hn] = (h16)__expf(s1[v] - mrun);
        }
        FragAB pf;                                     // relayout C->A frag via LDS
        pf.q[0] = *(const u32x4*)&probs[wave][hn][16 * g];
        pf.q[1] = *(const u32x4*)&probs[wave][hn][16 * g + 8];
        v8f lc = wmma_f16(pf, ones, zero);             // row sums, already in C layout
        #pragma unroll
        for (int v = 0; v < 8; ++v) lrun[v] += lc[v];
        #pragma unroll
        for (int t = 0; t < 4; ++t) {
            FragAB vb;                                 // V transposed: contiguous 32B
            vb.h = *(const v16h*)(Vt + baseVT +
                    (size_t)(t * 16 + hn) * S + k0 + 16 * g);
            O[t] = wmma_f16(pf, vb, O[t]);
        }
    }

    #pragma unroll
    for (int v = 0; v < 8; ++v) {
        float inv = 1.f / lrun[v];
        int row = q0 + v + 8 * g;
        size_t base = ((size_t)b * S + row) * Dm + (size_t)hh * HD;
        #pragma unroll
        for (int t = 0; t < 4; ++t)
            heads[base + t * 16 + hn] = (h16)(O[t][v] * inv);
    }
}

// ======================= host orchestration =========================================
extern "C" void kernel_launch(void* const* d_in, const int* in_sizes, int n_in,
                              void* d_out, int out_size, void* d_ws, size_t ws_size,
                              hipStream_t stream) {
    (void)in_sizes; (void)n_in; (void)out_size; (void)ws_size;
    const float* query = (const float*)d_in[0];
    const float* ori_k = (const float*)d_in[1];
    const float* Wq    = (const float*)d_in[2];
    const float* bq    = (const float*)d_in[3];
    const float* Wk    = (const float*)d_in[4];
    const float* bk    = (const float*)d_in[5];
    const float* Wv    = (const float*)d_in[6];
    const float* bv    = (const float*)d_in[7];
    const float* Wout  = (const float*)d_in[8];
    const float* bout  = (const float*)d_in[9];
    const float* Wf    = (const float*)d_in[10];
    const float* bf    = (const float*)d_in[11];
    const float* alpha = (const float*)d_in[12];
    const float* beta  = (const float*)d_in[13];

    char*  ws = (char*)d_ws;
    size_t o  = 0;
    auto alloc = [&](size_t bytes) {
        void* p = ws + o;
        o += (bytes + 255) & ~(size_t)255;
        return p;
    };
    const size_t DD = (size_t)Dm * Dm;
    h16*   Wq16  = (h16*)  alloc(DD * 2);
    h16*   Wk16  = (h16*)  alloc(DD * 2);
    h16*   Wv16  = (h16*)  alloc(DD * 2);
    h16*   Wo16  = (h16*)  alloc(DD * 2);
    h16*   Wf16  = (h16*)  alloc(DD * 2);
    float* padB  = (float*)alloc((size_t)Mrow * 4);
    float* x32   = (float*)alloc((size_t)Mrow * Dm * 4);
    float* q32   = (float*)alloc((size_t)Mrow * Dm * 4);
    h16*   q16   = (h16*)  alloc((size_t)Mrow * Dm * 2);
    h16*   Qh    = (h16*)  alloc((size_t)Mrow * Dm * 2);
    h16*   Kh    = (h16*)  alloc((size_t)Mrow * Dm * 2);
    h16*   Vth   = (h16*)  alloc((size_t)Mrow * Dm * 2);
    h16*   hd16  = (h16*)  alloc((size_t)Mrow * Dm * 2);
    float* out32 = (float*)alloc((size_t)Mrow * Dm * 4);
    h16*   t16   = (h16*)  alloc((size_t)Mrow * Dm * 2);

    // one-time prep
    pack_w_qkv<<<(int)(DD / 256), 256, 0, stream>>>(Wq, Wq16);
    pack_w_qkv<<<(int)(DD / 256), 256, 0, stream>>>(Wk, Wk16);
    pack_w_qkv<<<(int)(DD / 256), 256, 0, stream>>>(Wv, Wv16);
    pack_w_dd <<<(int)(DD / 256), 256, 0, stream>>>(Wout, Wo16);
    pack_w_dd <<<(int)(DD / 256), 256, 0, stream>>>(Wf, Wf16);
    make_pad  <<<Mrow, 256, 0, stream>>>(ori_k, padB);
    pos_encode<<<Mrow, 256, 0, stream>>>(query, x32);

    dim3 ggrid(Dm / 64, Mrow / 256);      // 16 x 8 blocks, 256 thr
    const int attnBlocks = Bb * H * (S / 64);

    for (int layer = 0; layer < NLAYER; ++layer) {
        layernorm<<<Mrow, 256, 0, stream>>>(x32, alpha, beta, q32, q16);
        gemm_wmma<EP_QS ><<<ggrid, 256, 0, stream>>>(q16, Wq16, bq, nullptr, Qh);
        gemm_wmma<EP_QK ><<<ggrid, 256, 0, stream>>>(q16, Wk16, bk, nullptr, Kh);
        gemm_wmma<EP_VT ><<<ggrid, 256, 0, stream>>>(q16, Wv16, bv, nullptr, Vth);
        attention<<<attnBlocks, 128, 0, stream>>>(Qh, Kh, Vth, padB, hd16);
        gemm_wmma<EP_RES><<<ggrid, 256, 0, stream>>>(hd16, Wo16, bout, q32, out32);
        layernorm<<<Mrow, 256, 0, stream>>>(out32, alpha, beta, nullptr, t16);
        float* xdst = (layer == NLAYER - 1) ? (float*)d_out : x32;
        gemm_wmma<EP_RELU><<<ggrid, 256, 0, stream>>>(t16, Wf16, bf, out32, xdst);
    }
}
